// global_attention_1_9569187135778
// MI455X (gfx1250) — compile-verified, matrix operand order checked by reading, not run on hardware
//
#include <hip/hip_runtime.h>

#define EDIM 512
#define HDIM 1024
#define NVOCAB 32000
#define TLIN 128
#define TLOUT 128
#define G3H (3 * HDIM)

typedef __attribute__((ext_vector_type(16))) __bf16 v16bf;
typedef __attribute__((ext_vector_type(8)))  __bf16 v8bf;
typedef __attribute__((ext_vector_type(8)))  float  v8f;

// ---------------- utility kernels ----------------

__global__ void k_zero_f32(float* __restrict__ p, int n) {
  int i = blockIdx.x * blockDim.x + threadIdx.x;
  if (i < n) p[i] = 0.f;
}

// gather embedding row seq[t] and cast fp32 -> bf16
__global__ void k_gather_cast(const float* __restrict__ emb,
                              const int* __restrict__ seq,
                              __bf16* __restrict__ out) {
  int t = blockIdx.x;
  long row = seq[t];
  const float* src = emb + row * (long)EDIM;
  __bf16* dst = out + (long)t * EDIM;
  for (int e = threadIdx.x; e < EDIM; e += blockDim.x)
    dst[e] = (__bf16)src[e];
}

// ---------------- WMMA GEMM, M=128 register-blocked ----------------
// C[128,N] (fp32) = A[128,K] (bf16 row-major, stride lda) * W[N,K]^T
//                   (W fp32 row-major, stride ldw, start col coloff) + bias[n]
// One wave per 16-wide N column; the wave keeps all 8 M-tiles (128 rows) in
// registers so each W fragment is loaded+converted ONCE and feeds 8 WMMAs.
// W (the big operand, e.g. 131MB lin_W) is streamed exactly once.
// All 8 A fragments are fetched into distinct registers up-front each K-chunk
// so the 8 WMMAs issue back-to-back against decreasing loadcnt waits instead
// of a full load round-trip per tile.
// Fragment layouts per ISA 7.12.2: lane m/n = lane%16; lanes<16 hold
// K {0..7,16..23}, lanes>=16 hold K {8..15,24..31}.
__global__ __launch_bounds__(256)
void k_gemm_bf16_m128(const __bf16* __restrict__ A, int lda,
                      const float* __restrict__ W, int ldw, int coloff,
                      const float* __restrict__ bias,
                      float* __restrict__ C,
                      int N, int K)
{
  const int lane = threadIdx.x & 31;
  const int wid  = threadIdx.x >> 5;
  const int tn   = blockIdx.x * 8 + wid;
  if (tn >= (N >> 4)) return;          // uniform per-wave exit: EXEC all-1 inside

  const int krow = lane & 15;
  const int koff = (lane >> 4) << 3;   // 0 or 8

  const __bf16* arow = A + (long)krow * lda + koff;
  const float*  wrow = W + (long)(tn * 16 + krow) * ldw + coloff + koff;

  v8f acc[8];
#pragma unroll
  for (int i = 0; i < 8; ++i) acc[i] = (v8f){};

  for (int kk = 0; kk < K; kk += 32) {
    // prefetch next W chunk while this one converts/computes
    if (kk + 128 < K) __builtin_prefetch(wrow + kk + 128, 0, 0);

    // issue ALL A-tile loads first (16 b128 in flight, distinct registers)
    v8bf alo[8], ahi[8];
#pragma unroll
    for (int tm = 0; tm < 8; ++tm) {
      const __bf16* ap = arow + (long)(tm * 16) * lda + kk;
      alo[tm] = *(const v8bf*)(ap);
      ahi[tm] = *(const v8bf*)(ap + 16);
    }

    // convert W fragment (fp32 -> bf16) while A loads are in flight
    v16bf b;
#pragma unroll
    for (int i = 0; i < 8; ++i) {
      b[i]     = (__bf16)wrow[kk + i];
      b[i + 8] = (__bf16)wrow[kk + 16 + i];
    }

    // 8 back-to-back WMMAs reusing the same W fragment
#pragma unroll
    for (int tm = 0; tm < 8; ++tm) {
      v16bf a;
#pragma unroll
      for (int i = 0; i < 8; ++i) { a[i] = alo[tm][i]; a[i + 8] = ahi[tm][i]; }
      acc[tm] = __builtin_amdgcn_wmma_f32_16x16x32_bf16(
          /*neg_a=*/false, a, /*neg_b=*/false, b,
          /*c_mod=*/(short)0, acc[tm], /*reuse_a=*/false, /*reuse_b=*/false);
    }
  }

  const int n  = tn * 16 + krow;
  const float bn = bias ? bias[n] : 0.f;
#pragma unroll
  for (int tm = 0; tm < 8; ++tm) {
    const int mb = tm * 16 + ((lane >> 4) << 3);
#pragma unroll
    for (int i = 0; i < 8; ++i)
      C[(long)(mb + i) * N + n] = acc[tm][i] + bn;
  }
}

// ---------------- recurrent GRU step (wave-per-output GEMV) ----------------
// mode 0 (encoder): gi_row precomputed (incl. bih); gh_g = dot(Wrec_g, vec)+bhh_g;
//                   h = (1-z)*n + z*h_prev   (vec == h_prev)
// mode 1 (decoder): gi_g = gi_row_g + dot(Wrec_g, vec); gh_g = bhh_g (h0==0);
//                   h = (1-z)*n
__global__ __launch_bounds__(256)
void k_gru_step(const float* __restrict__ Wrec, int ldw, int coloff,
                const float* __restrict__ vec,
                const float* __restrict__ gi_row,
                const float* __restrict__ bhh,
                const float* __restrict__ h_prev,
                float* __restrict__ h_out,
                int mode)
{
  const int j    = (blockIdx.x * blockDim.x + threadIdx.x) >> 5;
  const int lane = threadIdx.x & 31;
  if (j >= HDIM) return;

  float d[3];
#pragma unroll
  for (int g = 0; g < 3; ++g) {
    const float* wr = Wrec + (long)(j + g * HDIM) * ldw + coloff;
    float acc = 0.f;
    for (int k = lane; k < HDIM; k += 32) acc += wr[k] * vec[k];
#pragma unroll
    for (int off = 16; off; off >>= 1) acc += __shfl_down(acc, off, 32);
    d[g] = acc;
  }

  if (lane == 0) {
    float ir, iz, in_, hr, hz, hn, hp;
    if (mode == 0) {
      ir = gi_row[j]; iz = gi_row[j + HDIM]; in_ = gi_row[j + 2 * HDIM];
      hr = d[0] + bhh[j]; hz = d[1] + bhh[j + HDIM]; hn = d[2] + bhh[j + 2 * HDIM];
      hp = h_prev[j];
    } else {
      ir = gi_row[j] + d[0]; iz = gi_row[j + HDIM] + d[1]; in_ = gi_row[j + 2 * HDIM] + d[2];
      hr = bhh[j]; hz = bhh[j + HDIM]; hn = bhh[j + 2 * HDIM];
      hp = 0.f;
    }
    float r = 1.f / (1.f + expf(-(ir + hr)));
    float z = 1.f / (1.f + expf(-(iz + hz)));
    float n = tanhf(in_ + r * hn);
    h_out[j] = (1.f - z) * n + z * hp;
  }
}

// scores[t'] = dot(enc_states[t'], h)   (128 waves)
__global__ __launch_bounds__(256)
void k_scores(const float* __restrict__ enc_st, const float* __restrict__ h,
              float* __restrict__ scores) {
  const int t    = (blockIdx.x * blockDim.x + threadIdx.x) >> 5;
  const int lane = threadIdx.x & 31;
  if (t >= TLIN) return;
  const float* row = enc_st + (long)t * HDIM;
  float acc = 0.f;
  for (int k = lane; k < HDIM; k += 32) acc += row[k] * h[k];
#pragma unroll
  for (int off = 16; off; off >>= 1) acc += __shfl_down(acc, off, 32);
  if (lane == 0) scores[t] = acc;
}

// softmax over 128 scores, context[k] = sum_t alpha_t * enc_states[t][k]
__global__ __launch_bounds__(1024)
void k_softmax_ctx(const float* __restrict__ scores,
                   const float* __restrict__ enc_st,
                   float* __restrict__ ctx) {
  __shared__ float sa[TLIN];
  __shared__ float sinv;
  const int tid = threadIdx.x;
  if (tid < TLIN) sa[tid] = scores[tid];
  __syncthreads();
  if (tid == 0) {
    float m = sa[0];
    for (int i = 1; i < TLIN; ++i) m = fmaxf(m, sa[i]);
    float s = 0.f;
    for (int i = 0; i < TLIN; ++i) { float e = expf(sa[i] - m); sa[i] = e; s += e; }
    sinv = 1.f / s;
  }
  __syncthreads();
  const float inv = sinv;
  const int k = tid;                       // blockDim == HDIM
  float acc = 0.f;
  for (int t = 0; t < TLIN; ++t) acc += sa[t] * enc_st[(long)t * HDIM + k];
  ctx[k] = acc * inv;
}

// ht_bar[i] = tanh( wc[i, :H] . context + wc[i, H:] . h );  fp32 + bf16 copies
__global__ __launch_bounds__(256)
void k_htbar(const float* __restrict__ wc, const float* __restrict__ ctx,
             const float* __restrict__ h,
             float* __restrict__ out_f, __bf16* __restrict__ out_b) {
  const int i    = (blockIdx.x * blockDim.x + threadIdx.x) >> 5;
  const int lane = threadIdx.x & 31;
  if (i >= HDIM) return;
  const float* row = wc + (long)i * (2 * HDIM);
  float acc = 0.f;
  for (int k = lane; k < HDIM; k += 32)
    acc += row[k] * ctx[k] + row[HDIM + k] * h[k];
#pragma unroll
  for (int off = 16; off; off >>= 1) acc += __shfl_down(acc, off, 32);
  if (lane == 0) {
    float v = tanhf(acc);
    out_f[i] = v;
    out_b[i] = (__bf16)v;
  }
}

// ---------------- launcher ----------------

extern "C" void kernel_launch(void* const* d_in, const int* in_sizes, int n_in,
                              void* d_out, int out_size, void* d_ws, size_t ws_size,
                              hipStream_t stream) {
  (void)in_sizes; (void)n_in; (void)out_size; (void)ws_size;

  const int*   in_seq  = (const int*)  d_in[0];
  const int*   out_seq = (const int*)  d_in[1];
  const float* enc_emb = (const float*)d_in[2];
  const float* dec_emb = (const float*)d_in[3];
  const float* Wih_e   = (const float*)d_in[4];
  const float* Whh_e   = (const float*)d_in[5];
  const float* bih_e   = (const float*)d_in[6];
  const float* bhh_e   = (const float*)d_in[7];
  const float* Wih_d   = (const float*)d_in[8];
  /* Whh_d (d_in[9]) is multiplied by h0==0 in the reference -> unused */
  const float* bih_d   = (const float*)d_in[10];
  const float* bhh_d   = (const float*)d_in[11];
  const float* wc      = (const float*)d_in[12];
  const float* lin_W   = (const float*)d_in[13];
  const float* lin_b   = (const float*)d_in[14];
  float* out = (float*)d_out;

  char* w = (char*)d_ws;
  size_t o = 0;
  auto alloc = [&](size_t bytes) -> void* {
    void* p = w + o;
    o = (o + bytes + 255) & ~(size_t)255;
    return p;
  };
  __bf16* enc_x   = (__bf16*)alloc((size_t)TLIN  * EDIM * 2);
  __bf16* dec_x   = (__bf16*)alloc((size_t)TLOUT * EDIM * 2);
  float*  gi_e    = (float*) alloc((size_t)TLIN  * G3H  * 4);
  float*  gi_d    = (float*) alloc((size_t)TLOUT * G3H  * 4);
  float*  enc_st  = (float*) alloc((size_t)TLIN  * HDIM * 4);
  float*  htbar_f = (float*) alloc((size_t)TLOUT * HDIM * 4);
  __bf16* htbar_b = (__bf16*)alloc((size_t)TLOUT * HDIM * 2);
  float*  h_dec   = (float*) alloc((size_t)HDIM * 4);
  float*  scores  = (float*) alloc((size_t)TLIN * 4);
  float*  context = (float*) alloc((size_t)HDIM * 4);
  float*  zeros   = (float*) alloc((size_t)HDIM * 4);

  k_zero_f32<<<1, 1024, 0, stream>>>(zeros, HDIM);

  // embedding gathers (fp32 -> bf16 activations)
  k_gather_cast<<<TLIN,  256, 0, stream>>>(enc_emb, in_seq,  enc_x);
  k_gather_cast<<<TLOUT, 256, 0, stream>>>(dec_emb, out_seq, dec_x);

  // batched input projections: gi = X @ Wih^T + bih   (WMMA, M=128 blocked)
  {
    const int grid = (G3H / 16 + 7) / 8;              // 192 waves -> 24 blocks
    k_gemm_bf16_m128<<<grid, 256, 0, stream>>>(enc_x, EDIM, Wih_e, EDIM, 0,
                                               bih_e, gi_e, G3H, EDIM);
    k_gemm_bf16_m128<<<grid, 256, 0, stream>>>(dec_x, EDIM, Wih_d, EDIM + HDIM, 0,
                                               bih_d, gi_d, G3H, EDIM);
  }

  // encoder recurrence: enc_states rows double as the hidden-state chain
  for (int t = 0; t < TLIN; ++t) {
    const float* hp = t ? enc_st + (long)(t - 1) * HDIM : zeros;
    k_gru_step<<<128, 256, 0, stream>>>(Whh_e, HDIM, 0, hp,
                                        gi_e + (long)t * G3H, bhh_e, hp,
                                        enc_st + (long)t * HDIM, 0);
  }

  // decoder recurrence + attention; logits deferred to one big GEMM
  for (int t = 0; t < TLOUT; ++t) {
    const float* hb = t ? htbar_f + (long)(t - 1) * HDIM : zeros;
    k_gru_step<<<128, 256, 0, stream>>>(Wih_d, EDIM + HDIM, EDIM, hb,
                                        gi_d + (long)t * G3H, bhh_d, zeros,
                                        h_dec, 1);
    k_scores<<<16, 256, 0, stream>>>(enc_st, h_dec, scores);
    k_softmax_ctx<<<1, 1024, 0, stream>>>(scores, enc_st, context);
    k_htbar<<<128, 256, 0, stream>>>(wc, context, h_dec,
                                     htbar_f + (long)t * HDIM,
                                     htbar_b + (long)t * HDIM);
  }

  // logits for all steps: out[128, 32000] = htbar @ lin_W^T + lin_b
  // (WMMA, lin_W streamed exactly once)
  {
    const int grid = (NVOCAB / 16 + 7) / 8;           // 2000 waves -> 250 blocks
    k_gemm_bf16_m128<<<grid, 256, 0, stream>>>(htbar_b, HDIM, lin_W, HDIM, 0,
                                               lin_b, out, NVOCAB, HDIM);
  }
}
